// SimpleHeteroGAT_6047313952838
// MI455X (gfx1250) — compile-verified, matrix operand order checked by reading, not run on hardware
//
#include <hip/hip_runtime.h>
#include <hip/hip_bf16.h>
#include <math.h>

// ---------------------------------------------------------------------------
// SimpleHeteroGAT on MI455X (gfx1250, wave32, WMMA)
//
// Pipeline:
//   1) split W_proj / W_out into transposed bf16 hi/lo tables ([N][K])
//   2) h = x @ W + b  via v_wmma_f32_16x16x32_bf16, 3-WMMA hi/lo split (~fp32)
//   3) zero accumulators
//   4) 3 edge relations: gather-scale + scatter-add (global_atomic_add_f32)
//   5) out_u = elu(acc_u) @ W_out + b_out   (ELU fused into LDS staging)
//   6) out_i = elu(acc_i) in place in d_out
//
// Staging is fully vectorized: A tiles move as float4 global loads ->
// packed-bf16 uint2 LDS stores; B tiles move as uint4 (8 x bf16) b128
// copies. Fragments are read back as 2 x ds_load_b128 per operand.
// ---------------------------------------------------------------------------

typedef __attribute__((ext_vector_type(16))) __bf16 bfx16;
typedef __attribute__((ext_vector_type(8)))  __bf16 bfx8;
typedef __attribute__((ext_vector_type(8)))  float  fx8;

// Load one 16-element bf16 fragment for this lane from an LDS row of 32 K
// values. K mapping (ISA 7.12.2, 16-bit A 16x32): elements 0..7 -> K=8h+0..7,
// elements 8..15 -> K=16+8h+0..7  (h = lane>>4). Two ds_load_b128.
__device__ inline bfx16 load_frag32(const __bf16* rowBase, int h) {
    bfx8 c0 = *(const bfx8*)(rowBase + 8 * h);
    bfx8 c1 = *(const bfx8*)(rowBase + 16 + 8 * h);
    bfx16 f;
#pragma unroll
    for (int i = 0; i < 8; ++i) { f[i] = c0[i]; f[i + 8] = c1[i]; }
    return f;
}

__device__ inline fx8 wmma_bf16(bfx16 a, bfx16 b, fx8 c) {
    // (neg_a, A, neg_b, B, c_mod, C, reuse_a, reuse_b)
    return __builtin_amdgcn_wmma_f32_16x16x32_bf16(false, a, false, b,
                                                   (short)0, c, false, false);
}

__device__ inline void bf16_split(float x, __bf16& hi, __bf16& lo) {
    hi = (__bf16)x;
    lo = (__bf16)(x - (float)hi);
}

__device__ inline unsigned pack2(__bf16 a, __bf16 b) {
    unsigned short ua = __builtin_bit_cast(unsigned short, a);
    unsigned short ub = __builtin_bit_cast(unsigned short, b);
    return ((unsigned)ub << 16) | (unsigned)ua;
}

// Split a float4 into packed hi/lo bf16 uint2's.
__device__ inline void split4(float4 x, uint2& hiP, uint2& loP) {
    __bf16 h0, l0, h1, l1, h2, l2, h3, l3;
    bf16_split(x.x, h0, l0);
    bf16_split(x.y, h1, l1);
    bf16_split(x.z, h2, l2);
    bf16_split(x.w, h3, l3);
    hiP.x = pack2(h0, h1); hiP.y = pack2(h2, h3);
    loP.x = pack2(l0, l1); loP.y = pack2(l2, l3);
}

// --------------------------- prep: W -> Wt hi/lo ---------------------------
// W row-major [K][N]  ->  wtHi/wtLo [N][K] (bf16 hi/lo split)
__global__ void hgat_prep_wt(const float* __restrict__ W,
                             __bf16* __restrict__ wtHi,
                             __bf16* __restrict__ wtLo, int K, int N) {
    int e = blockIdx.x * blockDim.x + threadIdx.x;
    if (e >= K * N) return;
    int col = e % N, k = e / N;
    __bf16 h, l;
    bf16_split(W[k * N + col], h, l);
    wtHi[col * K + k] = h;
    wtLo[col * K + k] = l;
}

// ------------------------------- zero fill ---------------------------------
__global__ void hgat_zero4(float4* __restrict__ p, int n4) {
    int i = blockIdx.x * blockDim.x + threadIdx.x;
    if (i < n4) p[i] = make_float4(0.f, 0.f, 0.f, 0.f);
}

// --------------------- projection GEMM: H = X@W + b ------------------------
// X: [nRows][256] fp32, wt*: [128][256] bf16, H: [nRows][128] fp32.
// Block = 256 threads (8 waves) computes a 128x128 tile of H.
// Each wave owns a 16-row strip x 128 cols = 8 accumulator tiles (8 x v8f).
__global__ void __launch_bounds__(256)
hgat_gemm_proj(const float* __restrict__ X, const __bf16* __restrict__ wtHi,
               const __bf16* __restrict__ wtLo, const float* __restrict__ bias,
               float* __restrict__ H, int nRows) {
    __shared__ __align__(16) __bf16 sAh[128 * 32], sAl[128 * 32];
    __shared__ __align__(16) __bf16 sBh[128 * 32], sBl[128 * 32];

    const int tid  = threadIdx.x;
    const int wave = tid >> 5;
    const int lane = tid & 31;
    const int kh   = lane >> 4;   // lane half -> K block select
    const int lm   = lane & 15;   // row (A) / col (B) within tile
    const int r0   = blockIdx.x * 128;

    fx8 acc[8] = {};

    for (int kb = 0; kb < 256; kb += 32) {
        // stage A chunk: 128 rows x 32 K, float4 loads -> packed bf16 hi/lo
#pragma unroll
        for (int i = 0; i < 4; ++i) {
            int idx = i * 256 + tid;        // one float4 (4 elements)
            int row = idx >> 3;
            int k   = (idx & 7) * 4;
            int gr = r0 + row; if (gr >= nRows) gr = nRows - 1;
            float4 x = *(const float4*)&X[(size_t)gr * 256 + kb + k];
            uint2 hp, lp;
            split4(x, hp, lp);
            *(uint2*)&sAh[row * 32 + k] = hp;
            *(uint2*)&sAl[row * 32 + k] = lp;
        }
        // stage B chunk: 128 cols x 32 K, b128 copies from transposed tables
#pragma unroll
        for (int i = 0; i < 2; ++i) {
            int idx = i * 256 + tid;        // one uint4 (8 bf16)
            int col = idx >> 2;
            int k   = (idx & 3) * 8;
            *(uint4*)&sBh[col * 32 + k] = *(const uint4*)&wtHi[col * 256 + kb + k];
            *(uint4*)&sBl[col * 32 + k] = *(const uint4*)&wtLo[col * 256 + kb + k];
        }
        __syncthreads();

        const int arow = wave * 16 + lm;
        bfx16 aH = load_frag32(&sAh[arow * 32], kh);
        bfx16 aL = load_frag32(&sAl[arow * 32], kh);
#pragma unroll
        for (int t = 0; t < 8; ++t) {
            const int col = t * 16 + lm;
            bfx16 bH = load_frag32(&sBh[col * 32], kh);
            bfx16 bL = load_frag32(&sBl[col * 32], kh);
            acc[t] = wmma_bf16(aH, bH, acc[t]);
            acc[t] = wmma_bf16(aH, bL, acc[t]);
            acc[t] = wmma_bf16(aL, bH, acc[t]);
        }
        __syncthreads();
    }

    // store: lane l, element e -> (M = e + 8*kh, N = lm) + bias
#pragma unroll
    for (int t = 0; t < 8; ++t) {
        const int col = t * 16 + lm;
        const float b = bias[col];
#pragma unroll
        for (int e = 0; e < 8; ++e) {
            int gr = r0 + wave * 16 + e + 8 * kh;
            if (gr < nRows) H[(size_t)gr * 128 + col] = acc[t][e] + b;
        }
    }
}

// ------------------------- edge gather / scatter ---------------------------
// One wave per edge; lane covers 4 of the 128 features.
__global__ void __launch_bounds__(256)
hgat_edge_scatter(const float* __restrict__ Hsrc, const float* __restrict__ Hdst,
                  float* __restrict__ Out, const int* __restrict__ edges,
                  const float* __restrict__ alP, const float* __restrict__ arP,
                  int E) {
    int gid  = blockIdx.x * blockDim.x + threadIdx.x;
    int e    = gid >> 5;
    int lane = gid & 31;
    if (e >= E) return;
    const float al = alP[0], ar = arP[0];
    const int src = edges[2 * e + 0];
    const int dst = edges[2 * e + 1];
    const float4 s = *(const float4*)&Hsrc[(size_t)src * 128 + lane * 4];
    const float4 d = *(const float4*)&Hdst[(size_t)dst * 128 + lane * 4];
    float* o = &Out[(size_t)dst * 128 + lane * 4];
    unsafeAtomicAdd(o + 0, al * s.x + ar * d.x);
    unsafeAtomicAdd(o + 1, al * s.y + ar * d.y);
    unsafeAtomicAdd(o + 2, al * s.z + ar * d.z);
    unsafeAtomicAdd(o + 3, al * s.w + ar * d.w);
}

// ---------------- output GEMM: out_u = elu(acc) @ W_out + b ----------------
// ACC: [nRows][128] fp32, wt*: [64][128] bf16, OUT: [nRows][64] fp32.
// Block computes 128 rows x 64 cols; wave = 16-row strip x 4 tiles.
__global__ void __launch_bounds__(256)
hgat_gemm_out(const float* __restrict__ ACC, const __bf16* __restrict__ wtHi,
              const __bf16* __restrict__ wtLo, const float* __restrict__ bias,
              float* __restrict__ OUT, int nRows) {
    __shared__ __align__(16) __bf16 sAh[128 * 32], sAl[128 * 32];
    __shared__ __align__(16) __bf16 sBh[64 * 32],  sBl[64 * 32];

    const int tid  = threadIdx.x;
    const int wave = tid >> 5;
    const int lane = tid & 31;
    const int kh   = lane >> 4;
    const int lm   = lane & 15;
    const int r0   = blockIdx.x * 128;

    fx8 acc[4] = {};

    for (int kb = 0; kb < 128; kb += 32) {
        // stage A = elu(ACC chunk): float4 loads -> packed bf16 hi/lo
#pragma unroll
        for (int i = 0; i < 4; ++i) {
            int idx = i * 256 + tid;
            int row = idx >> 3;
            int k   = (idx & 7) * 4;
            int gr = r0 + row; if (gr >= nRows) gr = nRows - 1;
            float4 x = *(const float4*)&ACC[(size_t)gr * 128 + kb + k];
            x.x = (x.x > 0.f) ? x.x : expm1f(x.x);   // ELU fused here
            x.y = (x.y > 0.f) ? x.y : expm1f(x.y);
            x.z = (x.z > 0.f) ? x.z : expm1f(x.z);
            x.w = (x.w > 0.f) ? x.w : expm1f(x.w);
            uint2 hp, lp;
            split4(x, hp, lp);
            *(uint2*)&sAh[row * 32 + k] = hp;
            *(uint2*)&sAl[row * 32 + k] = lp;
        }
        // stage B: 64 cols x 32 K, one uint4 per thread
        {
            int idx = tid;
            int col = idx >> 2;
            int k   = (idx & 3) * 8;
            *(uint4*)&sBh[col * 32 + k] = *(const uint4*)&wtHi[col * 128 + kb + k];
            *(uint4*)&sBl[col * 32 + k] = *(const uint4*)&wtLo[col * 128 + kb + k];
        }
        __syncthreads();

        const int arow = wave * 16 + lm;
        bfx16 aH = load_frag32(&sAh[arow * 32], kh);
        bfx16 aL = load_frag32(&sAl[arow * 32], kh);
#pragma unroll
        for (int t = 0; t < 4; ++t) {
            const int col = t * 16 + lm;
            bfx16 bH = load_frag32(&sBh[col * 32], kh);
            bfx16 bL = load_frag32(&sBl[col * 32], kh);
            acc[t] = wmma_bf16(aH, bH, acc[t]);
            acc[t] = wmma_bf16(aH, bL, acc[t]);
            acc[t] = wmma_bf16(aL, bH, acc[t]);
        }
        __syncthreads();
    }

#pragma unroll
    for (int t = 0; t < 4; ++t) {
        const int col = t * 16 + lm;
        const float b = bias[col];
#pragma unroll
        for (int e = 0; e < 8; ++e) {
            int gr = r0 + wave * 16 + e + 8 * kh;
            if (gr < nRows) OUT[(size_t)gr * 64 + col] = acc[t][e] + b;
        }
    }
}

// -------------------------- in-place ELU (items) ---------------------------
__global__ void hgat_elu_inplace4(float4* __restrict__ p, int n4) {
    int i = blockIdx.x * blockDim.x + threadIdx.x;
    if (i < n4) {
        float4 x = p[i];
        x.x = (x.x > 0.f) ? x.x : expm1f(x.x);
        x.y = (x.y > 0.f) ? x.y : expm1f(x.y);
        x.z = (x.z > 0.f) ? x.z : expm1f(x.z);
        x.w = (x.w > 0.f) ? x.w : expm1f(x.w);
        p[i] = x;
    }
}

// ---------------------------------------------------------------------------
extern "C" void kernel_launch(void* const* d_in, const int* in_sizes, int n_in,
                              void* d_out, int out_size, void* d_ws, size_t ws_size,
                              hipStream_t stream) {
    const float* x_user = (const float*)d_in[0];
    const float* x_item = (const float*)d_in[1];
    const float* W_proj = (const float*)d_in[2];
    const float* b_proj = (const float*)d_in[3];
    const float* al_uu  = (const float*)d_in[4];
    const float* ar_uu  = (const float*)d_in[5];
    const float* al_iu  = (const float*)d_in[6];
    const float* ar_iu  = (const float*)d_in[7];
    const float* al_ui  = (const float*)d_in[8];
    const float* ar_ui  = (const float*)d_in[9];
    const float* W_out  = (const float*)d_in[10];
    const float* b_out  = (const float*)d_in[11];
    const int* edges_uu = (const int*)d_in[12];
    const int* edges_iu = (const int*)d_in[13];
    const int* edges_ui = (const int*)d_in[14];

    const int IN = 256, HID = 128, OUT = 64;
    const int N  = in_sizes[0] / IN;          // 50000 (users == items)
    const int E  = in_sizes[12] / 2;          // 200000

    // workspace layout
    char* ws = (char*)d_ws;
    const size_t hBytes = (size_t)N * HID * sizeof(float);
    float*  Hu   = (float*)(ws);
    float*  Hi   = (float*)(ws + hBytes);
    float*  AccU = (float*)(ws + 2 * hBytes);
    __bf16* wtpH = (__bf16*)(ws + 3 * hBytes);
    __bf16* wtpL = wtpH + (size_t)HID * IN;
    __bf16* wtoH = wtpL + (size_t)HID * IN;
    __bf16* wtoL = wtoH + (size_t)OUT * HID;

    float* outU = (float*)d_out;                       // [N][64]
    float* outI = (float*)d_out + (size_t)N * OUT;     // [N][128], accumulate here

    const int T = 256;
    const int gemmBlocks = (N + 127) / 128;
    const int edgeBlocks = (E + 7) / 8;                // 8 waves (edges) / block
    const int n4 = (N * HID) / 4;                      // HID multiple of 4

    // 1) weight prep
    hgat_prep_wt<<<(IN * HID + T - 1) / T, T, 0, stream>>>(W_proj, wtpH, wtpL, IN, HID);
    hgat_prep_wt<<<(HID * OUT + T - 1) / T, T, 0, stream>>>(W_out,  wtoH, wtoL, HID, OUT);

    // 2) projections (WMMA)
    hgat_gemm_proj<<<gemmBlocks, T, 0, stream>>>(x_user, wtpH, wtpL, b_proj, Hu, N);
    hgat_gemm_proj<<<gemmBlocks, T, 0, stream>>>(x_item, wtpH, wtpL, b_proj, Hi, N);

    // 3) zero accumulators (user acc in ws, item acc directly in d_out)
    hgat_zero4<<<(n4 + T - 1) / T, T, 0, stream>>>((float4*)AccU, n4);
    hgat_zero4<<<(n4 + T - 1) / T, T, 0, stream>>>((float4*)outI, n4);

    // 4) edge relations: gather-scale, scatter-add
    hgat_edge_scatter<<<edgeBlocks, T, 0, stream>>>(Hu, Hu, AccU, edges_uu, al_uu, ar_uu, E);
    hgat_edge_scatter<<<edgeBlocks, T, 0, stream>>>(Hi, Hu, AccU, edges_iu, al_iu, ar_iu, E);
    hgat_edge_scatter<<<edgeBlocks, T, 0, stream>>>(Hu, Hi, outI, edges_ui, al_ui, ar_ui, E);

    // 5) out_u = elu(AccU) @ W_out + b_out   (ELU fused, WMMA)
    hgat_gemm_out<<<gemmBlocks, T, 0, stream>>>(AccU, wtoH, wtoL, b_out, outU, N);

    // 6) out_i = elu(out_i) in place
    hgat_elu_inplace4<<<(n4 + T - 1) / T, T, 0, stream>>>((float4*)outI, n4);
}
